// Attn_34428457844860
// MI455X (gfx1250) — matrix operand order repaired and checked by reading, not compiled, not run on hardware
//
#include <hip/hip_runtime.h>
#include <cstdint>
#include <cstddef>

// ---------------------------------------------------------------------------
// CDNA5 (gfx1250) wave32 WMMA attention implementation.
// All matmuls routed through v_wmma_f32_16x16x32_f16 (f16 in, f32 accum).
// V is stored transposed (d-major) so PV B-fragments are contiguous b128 loads.
// ---------------------------------------------------------------------------

typedef _Float16 f16;
typedef __attribute__((ext_vector_type(16))) _Float16      v16h;
typedef __attribute__((ext_vector_type(8)))  float         v8f;
typedef __attribute__((ext_vector_type(4)))  unsigned int  u32x4;

union FragH {
    v16h  v;
    f16   h[16];
    u32x4 q[2];
};

__device__ __forceinline__ v8f wmma_f16(const v16h& a, const v16h& b, const v8f& c) {
    return __builtin_amdgcn_wmma_f32_16x16x32_f16(false, a, false, b, (short)0, c, false, false);
}

// problem constants
constexpr int cB   = 2;
constexpr int cT   = 2048;
constexpr int cD   = 2048;
constexpr int cNH  = 16;
constexpr int cNKV = 4;
constexpr int cHD  = 128;
constexpr int cKD  = cNKV * cHD;   // 512
constexpr float cEPS = 1.1920929e-7f;          // jnp.finfo(f32).eps
constexpr float cISQ = 0.08838834764831845f;   // 1/sqrt(128)

// ---------------------------------------------------------------------------
// fp32 -> f16 elementwise convert
// ---------------------------------------------------------------------------
__global__ void cvt_f32_f16(const float* __restrict__ in, f16* __restrict__ out, size_t n) {
    size_t i = (size_t)blockIdx.x * blockDim.x + threadIdx.x;
    if (i < n) out[i] = (f16)in[i];
}

// ---------------------------------------------------------------------------
// V transpose + convert: vf [(b*T+s)*NKV + kvh]*HD + d  (f32)
//                     -> vt [((b*NKV+kvh)*HD + d)*T + s] (f16, d-major)
// Output-coalesced; input gathers are L2-friendly (8 MB total).
// ---------------------------------------------------------------------------
__global__ void v_transpose(const float* __restrict__ vf, f16* __restrict__ vt, size_t n) {
    size_t o = (size_t)blockIdx.x * blockDim.x + threadIdx.x;
    if (o >= n) return;
    const int s   = (int)(o % cT);
    const int d   = (int)((o / cT) % cHD);
    const int kvh = (int)((o / ((size_t)cT * cHD)) % cNKV);
    const int b   = (int)(o / ((size_t)cT * cHD * cNKV));
    vt[o] = (f16)vf[((size_t)(b * cT + s) * cNKV + kvh) * cHD + d];
}

// ---------------------------------------------------------------------------
// GEMM (NT): C[M][N] = sum_k A[m][k] * B[n][k].  A: MxK row-major (f16),
// B: NxK row-major (f16), C: MxN row-major (f32).  M%128==0, N%128==0, K%32==0.
// Block: 256 threads (8 waves, 4 along M x 2 along N). Wave tile 32x64.
// ---------------------------------------------------------------------------
#define GM 128
#define GN 128
#define GK 32
#define LDH 48   // padded LDS row stride in halves (96B, multiple of 16B)

__global__ __launch_bounds__(256) void gemm_nt(
    const f16* __restrict__ A, const f16* __restrict__ Bm, float* __restrict__ C,
    int M, int N, int K)
{
    __shared__ f16 As[GM * LDH];
    __shared__ f16 Bs[GN * LDH];

    const int tid  = threadIdx.x;
    const int lane = tid & 31;
    const int wave = tid >> 5;
    const int wm   = wave & 3;          // 0..3 along M
    const int wn   = wave >> 2;         // 0..1 along N
    const int ln   = lane & 15;
    const int g    = lane >> 4;
    const int m0   = blockIdx.y * GM;
    const int n0   = blockIdx.x * GN;

    v8f acc[2][4];
    const v8f vzero = {0.f,0.f,0.f,0.f,0.f,0.f,0.f,0.f};
#pragma unroll
    for (int mt = 0; mt < 2; ++mt)
#pragma unroll
        for (int nt = 0; nt < 4; ++nt) acc[mt][nt] = vzero;

    const int lrow = tid >> 2;          // 0..63
    const int lcol = (tid & 3) * 8;     // 0,8,16,24

    for (int k0 = 0; k0 < K; k0 += GK) {
        // cooperative tile load: 16B per thread per pass, 2 passes per matrix
#pragma unroll
        for (int i = 0; i < 2; ++i) {
            const int r = lrow + i * 64;
            *(u32x4*)&As[r * LDH + lcol] =
                *(const u32x4*)&A[(size_t)(m0 + r) * K + k0 + lcol];
            *(u32x4*)&Bs[r * LDH + lcol] =
                *(const u32x4*)&Bm[(size_t)(n0 + r) * K + k0 + lcol];
        }
        __syncthreads();

        FragH af[2], bf[4];
#pragma unroll
        for (int mt = 0; mt < 2; ++mt) {
            const int m = wm * 32 + mt * 16 + ln;
            af[mt].q[0] = *(const u32x4*)&As[m * LDH + 8 * g];        // K in [8g, 8g+8)
            af[mt].q[1] = *(const u32x4*)&As[m * LDH + 16 + 8 * g];   // K in [16+8g, 24+8g)
        }
#pragma unroll
        for (int nt = 0; nt < 4; ++nt) {
            const int n = wn * 64 + nt * 16 + ln;
            bf[nt].q[0] = *(const u32x4*)&Bs[n * LDH + 8 * g];        // K in [8g, 8g+8)
            bf[nt].q[1] = *(const u32x4*)&Bs[n * LDH + 16 + 8 * g];
        }
#pragma unroll
        for (int mt = 0; mt < 2; ++mt)
#pragma unroll
            for (int nt = 0; nt < 4; ++nt)
                acc[mt][nt] = wmma_f16(af[mt].v, bf[nt].v, acc[mt][nt]);
        __syncthreads();
    }

    // epilogue: D layout -> lane ln holds N=ln, VGPR r holds M=r+8g
#pragma unroll
    for (int mt = 0; mt < 2; ++mt)
#pragma unroll
        for (int nt = 0; nt < 4; ++nt) {
            const int cn = n0 + wn * 64 + nt * 16 + ln;
#pragma unroll
            for (int r = 0; r < 8; ++r) {
                const int cm = m0 + wm * 32 + mt * 16 + r + 8 * g;
                C[(size_t)cm * N + cn] = acc[mt][nt][r];
            }
        }
}

// ---------------------------------------------------------------------------
// q/k post-processing: RMS norm + partial RoPE (first 16 dims) + optional gain.
// One wave per head-row of 128 floats.  rid = (b*T + t)*nheads + h.
// ---------------------------------------------------------------------------
__global__ __launch_bounds__(32) void qk_postproc(
    const float* __restrict__ X, f16* __restrict__ Y,
    int nheads, const float* __restrict__ gains, int apply_gain)
{
    const int rid  = blockIdx.x;
    const int h    = rid % nheads;
    const int t    = (rid / nheads) % cT;
    const int lane = threadIdx.x;

    const float* xr = X + (size_t)rid * cHD + lane * 4;
    float x0 = xr[0], x1 = xr[1], x2 = xr[2], x3 = xr[3];

    float ss = x0*x0 + x1*x1 + x2*x2 + x3*x3;
#pragma unroll
    for (int off = 1; off < 32; off <<= 1) ss += __shfl_xor(ss, off);
    const float inv = rsqrtf(ss * (1.0f / 128.0f) + cEPS);
    x0 *= inv; x1 *= inv; x2 *= inv; x3 *= inv;

    // partial RoPE on dims 0..15 (lanes 0..3); shfl_xor(.,2) pairs d <-> d^8
    float p0 = __shfl_xor(x0, 2), p1 = __shfl_xor(x1, 2);
    float p2 = __shfl_xor(x2, 2), p3 = __shfl_xor(x3, 2);
    if (lane < 4) {
        float xin[4] = {x0, x1, x2, x3};
        float pin[4] = {p0, p1, p2, p3};
        float outv[4];
#pragma unroll
        for (int c = 0; c < 4; ++c) {
            const int d = lane * 4 + c;
            const int i = (d < 8) ? d : (d - 8);
            const float f = (float)t * __powf(10000.0f, -(float)i * 0.125f);
            const float cf = __cosf(f), sf = __sinf(f);
            outv[c] = (d < 8) ? (xin[c] * cf + pin[c] * sf)
                              : (-xin[c] * cf + pin[c] * sf);
        }
        x0 = outv[0]; x1 = outv[1]; x2 = outv[2]; x3 = outv[3];
    }

    float gsc = 1.0f;
    if (apply_gain) gsc = gains[h] * cISQ;   // fold qg[h] and 1/sqrt(HD) into q
    x0 *= gsc; x1 *= gsc; x2 *= gsc; x3 *= gsc;

    f16* yr = Y + (size_t)rid * cHD + lane * 4;
    yr[0] = (f16)x0; yr[1] = (f16)x1; yr[2] = (f16)x2; yr[3] = (f16)x3;
}

// ---------------------------------------------------------------------------
// Causal flash attention with GQA.  One wave per (16-row Q tile, head, batch).
// Q: [(b*T+t)*NH + h]*HD f16 ; K: [(b*T+s)*NKV + kvh]*HD f16 ;
// Vt: [((b*NKV+kvh)*HD + d)*T + s] f16 (transposed) ; O same layout as Q.
// S = Q K^T via 4 WMMA k-steps per 16x16 tile (K frags straight from global);
// online softmax; P re-laid out via 1KB LDS; O += P V via 8 WMMAs per chunk
// with Vt fragments as contiguous global b128 loads.
// ---------------------------------------------------------------------------
__global__ __launch_bounds__(32) void flash_attn(
    const f16* __restrict__ Q, const f16* __restrict__ Kk,
    const f16* __restrict__ Vt, f16* __restrict__ O)
{
    __shared__ f16 Ps[16 * 32];

    const int lane = threadIdx.x;
    const int ln   = lane & 15;
    const int g    = lane >> 4;
    const int qt   = blockIdx.x;
    const int h    = blockIdx.y;
    const int b    = blockIdx.z;
    const int kvh  = h >> 2;           // G = NH/NKV = 4

    // preload Q fragments for the whole 16x128 tile (A layout, 4 k-steps)
    FragH aq[4];
    {
        const int m = qt * 16 + ln;
        const f16* qrow = Q + ((size_t)(b * cT + m) * cNH + h) * cHD;
#pragma unroll
        for (int kk = 0; kk < 4; ++kk) {
            aq[kk].q[0] = *(const u32x4*)&qrow[32 * kk + 8 * g];
            aq[kk].q[1] = *(const u32x4*)&qrow[32 * kk + 16 + 8 * g];
        }
    }

    const f16* vbase = Vt + (size_t)(b * cNKV + kvh) * cHD * cT;  // + d*T + s

    float mrow[8], srow[8];
    v8f o[8];
    const v8f vzero = {0.f,0.f,0.f,0.f,0.f,0.f,0.f,0.f};
#pragma unroll
    for (int r = 0; r < 8; ++r) { mrow[r] = -1e30f; srow[r] = 0.f; }
#pragma unroll
    for (int t2 = 0; t2 < 8; ++t2) o[t2] = vzero;

    const int nchunks = (qt * 16 + 15) / 32 + 1;
    for (int sc = 0; sc < nchunks; ++sc) {
        const int s0 = sc * 32;

        // prefetch next chunk's K rows and V columns (global_prefetch_b8)
        if (sc + 1 < nchunks) {
            __builtin_prefetch(Kk + ((size_t)(b * cT + s0 + 32 + ln) * cNKV + kvh) * cHD, 0, 1);
            __builtin_prefetch(Kk + ((size_t)(b * cT + s0 + 48 + ln) * cNKV + kvh) * cHD, 0, 1);
            __builtin_prefetch(vbase + (size_t)lane * cT + s0 + 32, 0, 1);
        }

        // S tiles: keys [s0, s0+16) and [s0+16, s0+32); K frags from global
        v8f sT0 = vzero, sT1 = vzero;
        {
            const f16* krow0 = Kk + ((size_t)(b * cT + s0 + ln) * cNKV + kvh) * cHD;
            const f16* krow1 = Kk + ((size_t)(b * cT + s0 + 16 + ln) * cNKV + kvh) * cHD;
            FragH bk;
#pragma unroll
            for (int kk = 0; kk < 4; ++kk) {
                bk.q[0] = *(const u32x4*)&krow0[32 * kk + 8 * g];
                bk.q[1] = *(const u32x4*)&krow0[32 * kk + 16 + 8 * g];
                sT0 = wmma_f16(aq[kk].v, bk.v, sT0);
            }
#pragma unroll
            for (int kk = 0; kk < 4; ++kk) {
                bk.q[0] = *(const u32x4*)&krow1[32 * kk + 8 * g];
                bk.q[1] = *(const u32x4*)&krow1[32 * kk + 16 + 8 * g];
                sT1 = wmma_f16(aq[kk].v, bk.v, sT1);
            }
        }

        __threadfence_block();   // WAR: previous iteration's Ps reads done

        // online softmax over the 32 keys (per-row reduce across 16-lane halves)
#pragma unroll
        for (int r = 0; r < 8; ++r) {
            const int mg = qt * 16 + r + 8 * g;
            float v0 = ((s0 + ln)      <= mg) ? sT0[r] : -1e30f;
            float v1 = ((s0 + 16 + ln) <= mg) ? sT1[r] : -1e30f;
            float tmax = fmaxf(v0, v1);
#pragma unroll
            for (int off = 1; off < 16; off <<= 1)
                tmax = fmaxf(tmax, __shfl_xor(tmax, off));
            const float nm = fmaxf(mrow[r], tmax);
            const float rs = __expf(mrow[r] - nm);
            mrow[r] = nm;
            const float pe0 = __expf(v0 - nm);
            const float pe1 = __expf(v1 - nm);
            float psum = pe0 + pe1;
#pragma unroll
            for (int off = 1; off < 16; off <<= 1)
                psum += __shfl_xor(psum, off);
            srow[r] = srow[r] * rs + psum;
#pragma unroll
            for (int t2 = 0; t2 < 8; ++t2) o[t2][r] *= rs;
            Ps[(r + 8 * g) * 32 + ln]      = (f16)pe0;
            Ps[(r + 8 * g) * 32 + 16 + ln] = (f16)pe1;
        }
        __threadfence_block();   // LDS stores visible before cross-lane reads

        // P as A fragment (16x32); Vt column fragments straight from global
        FragH pa;
        pa.q[0] = *(const u32x4*)&Ps[ln * 32 + 8 * g];
        pa.q[1] = *(const u32x4*)&Ps[ln * 32 + 16 + 8 * g];
#pragma unroll
        for (int t2 = 0; t2 < 8; ++t2) {
            // B-frag: lane ln = output dim d = 16*t2+ln, elems = s0+8g..+8, s0+16+8g..+8
            const f16* vcol = vbase + (size_t)(16 * t2 + ln) * cT + s0 + 8 * g;
            FragH vb;
            vb.q[0] = *(const u32x4*)&vcol[0];
            vb.q[1] = *(const u32x4*)&vcol[16];
            o[t2] = wmma_f16(pa.v, vb.v, o[t2]);
        }
    }

    // normalize and store f16 output (same layout as Q)
#pragma unroll
    for (int t2 = 0; t2 < 8; ++t2) {
#pragma unroll
        for (int r = 0; r < 8; ++r) {
            const int m = qt * 16 + r + 8 * g;
            O[((size_t)(b * cT + m) * cNH + h) * cHD + 16 * t2 + ln] =
                (f16)(o[t2][r] / srow[r]);
        }
    }
}

// ---------------------------------------------------------------------------
// host
// ---------------------------------------------------------------------------
extern "C" void kernel_launch(void* const* d_in, const int* in_sizes, int n_in,
                              void* d_out, int out_size, void* d_ws, size_t ws_size,
                              hipStream_t stream) {
    (void)in_sizes; (void)n_in; (void)out_size; (void)ws_size;

    const float* x  = (const float*)d_in[0];
    const float* wq = (const float*)d_in[1];
    const float* wk = (const float*)d_in[2];
    const float* wv = (const float*)d_in[3];
    const float* wp = (const float*)d_in[4];
    const float* qg = (const float*)d_in[5];

    const size_t nX  = (size_t)cB * cT * cD;    // 8.39M
    const size_t nWq = (size_t)cD * cD;         // 4.19M
    const size_t nWk = (size_t)cKD * cD;        // 1.05M
    const size_t nK  = (size_t)cB * cT * cKD;   // 2.10M
    const int    M   = cB * cT;                 // 4096

    f16*   xh  = (f16*)d_ws;
    f16*   wqh = xh  + nX;
    f16*   wkh = wqh + nWq;
    f16*   wvh = wkh + nWk;
    f16*   wph = wvh + nWk;
    float* qf  = (float*)(wph + nWq);
    float* kf  = qf + nX;
    float* vf  = kf + nK;
    f16*   qh  = (f16*)(vf + nK);
    f16*   kh  = qh + nX;
    f16*   vth = kh + nK;                       // transposed V (f16)
    f16*   yh  = vth + nK;                      // total ~124 MB

    // 1) convert inputs to f16
    cvt_f32_f16<<<(int)((nX  + 255) / 256), 256, 0, stream>>>(x,  xh,  nX);
    cvt_f32_f16<<<(int)((nWq + 255) / 256), 256, 0, stream>>>(wq, wqh, nWq);
    cvt_f32_f16<<<(int)((nWk + 255) / 256), 256, 0, stream>>>(wk, wkh, nWk);
    cvt_f32_f16<<<(int)((nWk + 255) / 256), 256, 0, stream>>>(wv, wvh, nWk);
    cvt_f32_f16<<<(int)((nWq + 255) / 256), 256, 0, stream>>>(wp, wph, nWq);

    // 2) QKV projections (x @ W^T) via WMMA
    gemm_nt<<<dim3(cD  / GN, M / GM), 256, 0, stream>>>(xh, wqh, qf, M, cD,  cD);
    gemm_nt<<<dim3(cKD / GN, M / GM), 256, 0, stream>>>(xh, wkh, kf, M, cKD, cD);
    gemm_nt<<<dim3(cKD / GN, M / GM), 256, 0, stream>>>(xh, wvh, vf, M, cKD, cD);

    // 3) RMS norm + RoPE (+ qg*scale folded into q); v converted + transposed
    qk_postproc<<<cB * cT * cNH,  32, 0, stream>>>(qf, qh, cNH,  qg,      1);
    qk_postproc<<<cB * cT * cNKV, 32, 0, stream>>>(kf, kh, cNKV, nullptr, 0);
    v_transpose<<<(int)((nK + 255) / 256), 256, 0, stream>>>(vf, vth, nK);

    // 4) causal flash attention (GQA)
    flash_attn<<<dim3(cT / 16, cNH, cB), 32, 0, stream>>>(qh, kh, vth, yh);

    // 5) output projection -> fp32 result
    gemm_nt<<<dim3(cD / GN, M / GM), 256, 0, stream>>>(yh, wph, (float*)d_out, M, cD, cD);
}
